// StreamingTransformerLayer_28776280883458
// MI455X (gfx1250) — matrix operand order repaired and checked by reading, not compiled, not run on hardware
//
#include <hip/hip_runtime.h>
#include <hip/hip_bf16.h>

// ---------------------------------------------------------------------------
// Types for CDNA5 WMMA
// ---------------------------------------------------------------------------
typedef __bf16        v16bf __attribute__((ext_vector_type(16)));
typedef float         v8f   __attribute__((ext_vector_type(8)));
typedef unsigned int  u32x4 __attribute__((ext_vector_type(4)));
typedef int           v4i   __attribute__((vector_size(16)));

typedef __attribute__((address_space(1))) v4i as1_v4i;
typedef __attribute__((address_space(3))) v4i as3_v4i;

union Frag16 { u32x4 u[2]; v16bf v; };

__device__ __forceinline__ unsigned short f2bf(float f) {
    unsigned int u = __builtin_bit_cast(unsigned int, f);
    unsigned int r = u + 0x7FFFu + ((u >> 16) & 1u);   // round-to-nearest-even
    return (unsigned short)(r >> 16);
}
__device__ __forceinline__ float bf2f(unsigned short h) {
    unsigned int u = ((unsigned int)h) << 16;
    return __builtin_bit_cast(float, u);
}

// Problem constants (match reference setup_inputs)
#define NB   8
#define NT   256
#define NC   2048
#define NH   16
#define NDH  128
#define NCAP 2048
#define NFF  8192

// ---------------------------------------------------------------------------
// CDNA5 async global->LDS path (ASYNCcnt), with safe fallback
// ---------------------------------------------------------------------------
#if defined(__has_builtin)
#  if __has_builtin(__builtin_amdgcn_global_load_async_to_lds_b128)
#    define HAVE_ASYNC_LDS 1
#  endif
#  if __has_builtin(__builtin_amdgcn_s_wait_asynccnt)
#    define HAVE_WAIT_ASYNC 1
#  endif
#endif
#ifndef HAVE_ASYNC_LDS
#  define HAVE_ASYNC_LDS 0
#endif
#ifndef HAVE_WAIT_ASYNC
#  define HAVE_WAIT_ASYNC 0
#endif

__device__ __forceinline__ void async_copy_b128(const unsigned short* g, unsigned short* l) {
#if HAVE_ASYNC_LDS
    __builtin_amdgcn_global_load_async_to_lds_b128(
        (as1_v4i*)(void*)g,
        (as3_v4i*)(void*)l, 0, 0);
#else
    *(u32x4*)l = *(const u32x4*)g;           // global_load_b128 + ds_store_b128
#endif
}

__device__ __forceinline__ void wait_async_lds() {
#if HAVE_ASYNC_LDS
#  if HAVE_WAIT_ASYNC
    __builtin_amdgcn_s_wait_asynccnt(0);
#  else
    asm volatile("s_wait_asynccnt 0x0" ::: "memory");
#  endif
#endif
}

// ---------------------------------------------------------------------------
// Elementwise fp32 -> bf16 cast (weights)
// ---------------------------------------------------------------------------
__global__ __launch_bounds__(256) void cast_f32_bf16(const float* __restrict__ in,
                                                     unsigned short* __restrict__ out,
                                                     long long n) {
    long long i = (long long)blockIdx.x * 256 + threadIdx.x;
    long long stride = (long long)gridDim.x * 256;
    for (; i < n; i += stride) out[i] = f2bf(in[i]);
}

// ---------------------------------------------------------------------------
// RMSNorm (row length NC) -> bf16
// ---------------------------------------------------------------------------
__global__ __launch_bounds__(256) void rmsnorm_bf16(const float* __restrict__ x,
                                                    const float* __restrict__ alpha,
                                                    unsigned short* __restrict__ out) {
    __shared__ float red[256];
    const int row = blockIdx.x;
    const float* xr = x + (size_t)row * NC;
    float ss = 0.f;
    for (int c = threadIdx.x; c < NC; c += 256) { float v = xr[c]; ss += v * v; }
    red[threadIdx.x] = ss; __syncthreads();
    for (int s = 128; s > 0; s >>= 1) {
        if (threadIdx.x < s) red[threadIdx.x] += red[threadIdx.x + s];
        __syncthreads();
    }
    float inv = rsqrtf(red[0] * (1.0f / NC) + 1e-5f);
    for (int c = threadIdx.x; c < NC; c += 256)
        out[(size_t)row * NC + c] = f2bf(xr[c] * alpha[c] * inv);
}

// ---------------------------------------------------------------------------
// Generic bf16 WMMA GEMM:  C[M,N] = A[M,K] * B[N,K]^T   (both K-contiguous)
// Block tile 128x128, 4 waves (2x2), wave tile 64x64.
// K is consumed in 32-wide slabs double-buffered through LDS via async copies.
// Batched via blockIdx.z with element strides sA/sB/sC.
// ---------------------------------------------------------------------------
enum { EPI_F32 = 0, EPI_BF16 = 1, EPI_GELU_BF16 = 2, EPI_F32_RES = 3 };

#define KSTEP 32
#define LPAD  40   // LDS row stride in elements (80 B: 16B-aligned, conflict-free)

template <int EPI>
__global__ __launch_bounds__(128)
void gemm_bf16(const unsigned short* __restrict__ A,
               const unsigned short* __restrict__ Bm,
               void* __restrict__ Cout,
               const float* __restrict__ Res,
               int M, int N, int K,
               long long sA, long long sB, long long sC) {
    __shared__ __align__(16) unsigned short tA[2][128 * LPAD];
    __shared__ __align__(16) unsigned short tB[2][128 * LPAD];

    const int lane = threadIdx.x & 31;
    const int wave = threadIdx.x >> 5;
    const int lrow = lane & 15;
    const int half = lane >> 4;
    const int wm   = wave & 1;   // 2 waves along M
    const int wn   = wave >> 1;  // 2 waves along N

    const unsigned short* Ab = A  + (size_t)blockIdx.z * sA;
    const unsigned short* Bb = Bm + (size_t)blockIdx.z * sB;

    const int blockM = blockIdx.y * 128;
    const int blockN = blockIdx.x * 128;

    v8f acc[4][4] = {};

    // Stage one 128x32 slab of A and B into LDS buffer `buf` (async).
    auto stage = [&](int buf, int k) {
        #pragma unroll
        for (int it = 0; it < 4; ++it) {
            const int c   = threadIdx.x + it * 128;  // 512 16B-chunks per tile
            const int row = c >> 2;
            const int ch  = (c & 3) * 8;
            async_copy_b128(Ab + (size_t)(blockM + row) * K + k + ch,
                            &tA[buf][row * LPAD + ch]);
            async_copy_b128(Bb + (size_t)(blockN + row) * K + k + ch,
                            &tB[buf][row * LPAD + ch]);
        }
        if (k + 4 * KSTEP < K) {  // emits global_prefetch_b8
            __builtin_prefetch(Ab + (size_t)(blockM + (threadIdx.x & 127)) * K + k + 4 * KSTEP, 0, 0);
            __builtin_prefetch(Bb + (size_t)(blockN + (threadIdx.x & 127)) * K + k + 4 * KSTEP, 0, 0);
        }
    };

    stage(0, 0);
    wait_async_lds();
    __syncthreads();

    int buf = 0;
    for (int k = 0; k < K; k += KSTEP) {
        if (k + KSTEP < K) stage(buf ^ 1, k + KSTEP);  // overlap next slab

        Frag16 a[4], b[4];
        // A-matrix 16x32 bf16 layout: low half-wave K {0..7,16..23}, high {8..15,24..31}
        #pragma unroll
        for (int i = 0; i < 4; ++i) {
            const unsigned short* p = &tA[buf][(wm * 64 + i * 16 + lrow) * LPAD + half * 8];
            a[i].u[0] = *(const u32x4*)(p);
            a[i].u[1] = *(const u32x4*)(p + 16);
        }
        // B-matrix 32x16 bf16 layout: low half-wave K {0..7,16..23}, high {8..15,24..31}
        #pragma unroll
        for (int j = 0; j < 4; ++j) {
            const unsigned short* p = &tB[buf][(wn * 64 + j * 16 + lrow) * LPAD + half * 8];
            b[j].u[0] = *(const u32x4*)(p);
            b[j].u[1] = *(const u32x4*)(p + 16);
        }
        #pragma unroll
        for (int i = 0; i < 4; ++i)
            #pragma unroll
            for (int j = 0; j < 4; ++j)
                acc[i][j] = __builtin_amdgcn_wmma_f32_16x16x32_bf16(
                    false, a[i].v, false, b[j].v, (short)0, acc[i][j], false, false);

        wait_async_lds();
        __syncthreads();
        buf ^= 1;
    }

    // C/D 16x16 f32 layout: lane -> N = lane%16, VGPR r -> M = r + (lane>=16 ? 8 : 0)
    #pragma unroll
    for (int i = 0; i < 4; ++i) {
        #pragma unroll
        for (int j = 0; j < 4; ++j) {
            const int col = blockN + wn * 64 + j * 16 + lrow;
            #pragma unroll
            for (int r = 0; r < 8; ++r) {
                const int row = blockM + wm * 64 + i * 16 + r + half * 8;
                const size_t idx = (size_t)blockIdx.z * sC + (size_t)row * N + col;
                const float v = acc[i][j][r];
                if (EPI == EPI_F32) {
                    ((float*)Cout)[idx] = v;
                } else if (EPI == EPI_F32_RES) {
                    ((float*)Cout)[idx] = v + Res[(size_t)row * N + col];
                } else if (EPI == EPI_BF16) {
                    ((unsigned short*)Cout)[idx] = f2bf(v);
                } else { // exact GELU: 0.5*x*(1+erf(x/sqrt(2)))
                    const float g = 0.5f * v * (1.0f + erff(v * 0.70710678118654752f));
                    ((unsigned short*)Cout)[idx] = f2bf(g);
                }
            }
        }
    }
}

// ---------------------------------------------------------------------------
// RoPE + head split: qkv fp32 [B,T,3C] -> q/k/v bf16 [B,H,T,Dh] (q,k roped)
// One thread per (b,h,t,pair)  pair in [0,64)
// ---------------------------------------------------------------------------
__global__ __launch_bounds__(256) void rope_split(const float* __restrict__ qkv,
                                                  const int* __restrict__ offp,
                                                  unsigned short* __restrict__ qb,
                                                  unsigned short* __restrict__ kb,
                                                  unsigned short* __restrict__ vb) {
    const int off = *offp;
    const int idx = blockIdx.x * 256 + threadIdx.x;      // B*H*T*64
    const int i = idx & 63;
    const int t = (idx >> 6) & 255;
    const int h = (idx >> 14) & 15;
    const int b = idx >> 18;
    const float* row = qkv + (size_t)(b * NT + t) * (3 * NC);
    const float inv_freq = __expf(-(float)i * (logf(10000.0f) / 64.0f));
    const float ang = (float)(off + t) * inv_freq;
    const float c = cosf(ang), s = sinf(ang);
    const size_t o = (((size_t)(b * NH + h) * NT + t) * NDH) + 2 * i;
    {
        const float x1 = row[h * NDH + 2 * i], x2 = row[h * NDH + 2 * i + 1];
        qb[o] = f2bf(x1 * c - x2 * s); qb[o + 1] = f2bf(x1 * s + x2 * c);
    }
    {
        const float x1 = row[NC + h * NDH + 2 * i], x2 = row[NC + h * NDH + 2 * i + 1];
        kb[o] = f2bf(x1 * c - x2 * s); kb[o + 1] = f2bf(x1 * s + x2 * c);
    }
    vb[o]     = f2bf(row[2 * NC + h * NDH + 2 * i]);
    vb[o + 1] = f2bf(row[2 * NC + h * NDH + 2 * i + 1]);
}

// ---------------------------------------------------------------------------
// Merge ring-buffer cache updates: build bf16 K-cache [B,H,CAP,Dh] and
// transposed bf16 V-cache [B,H,Dh,CAP].
// ---------------------------------------------------------------------------
__global__ __launch_bounds__(256) void merge_cache(const float* __restrict__ kc,
                                                   const float* __restrict__ vc,
                                                   const unsigned short* __restrict__ knew,
                                                   const unsigned short* __restrict__ vnew,
                                                   const int* __restrict__ offp,
                                                   unsigned short* __restrict__ kout,
                                                   unsigned short* __restrict__ vTout) {
    const int off = *offp;
    const long long idx = (long long)blockIdx.x * 256 + threadIdx.x;  // B*H*CAP*Dh
    const int d = (int)(idx & 127);
    const int s = (int)((idx >> 7) & 2047);
    const int h = (int)((idx >> 18) & 15);
    const int b = (int)(idx >> 22);
    int u = (s - off) % NCAP; if (u < 0) u += NCAP;
    unsigned short kv, vv;
    if (u < NT) {
        const size_t nidx = (((size_t)(b * NH + h) * NT + u) * NDH) + d;
        kv = knew[nidx]; vv = vnew[nidx];
    } else {
        kv = f2bf(kc[idx]); vv = f2bf(vc[idx]);
    }
    kout[idx] = kv;
    vTout[(((size_t)(b * NH + h) * NDH + d) * NCAP) + s] = vv;
}

// ---------------------------------------------------------------------------
// Masked softmax over cache slots, in place on bf16 scores [B,H,T,CAP]
// (scale folded in; mask reproduces the reference ring-buffer position logic)
// ---------------------------------------------------------------------------
__global__ __launch_bounds__(256) void softmax_mask(unsigned short* __restrict__ attn,
                                                    const int* __restrict__ offp) {
    __shared__ float red[256];
    const int off = *offp;
    const int row = blockIdx.x;           // (b*H + h)*T + t
    const int t = row & (NT - 1);
    unsigned short* pr = attn + (size_t)row * NCAP;
    const int last = off + NT - 1;
    const int end_index = last % NCAP;
    const int qpos = off + t;
    const float scale = 0.08838834764831845f;   // 1/sqrt(128)
    float l[8];
    float mx = -1e30f;
    #pragma unroll
    for (int j = 0; j < 8; ++j) {
        const int s = threadIdx.x + j * 256;
        const int delta = s - end_index;
        int pos = (delta <= 0) ? (last + delta) : (last + delta - NCAP);
        if (s >= off + NT) pos = -1;
        const int d = qpos - pos;
        const bool ok = (pos >= 0) && (d >= 0) && (d < NCAP);
        l[j] = ok ? bf2f(pr[s]) * scale : -1e9f;
        mx = fmaxf(mx, l[j]);
    }
    red[threadIdx.x] = mx; __syncthreads();
    for (int st = 128; st > 0; st >>= 1) {
        if (threadIdx.x < st) red[threadIdx.x] = fmaxf(red[threadIdx.x], red[threadIdx.x + st]);
        __syncthreads();
    }
    mx = red[0]; __syncthreads();
    float sum = 0.f;
    #pragma unroll
    for (int j = 0; j < 8; ++j) { l[j] = __expf(l[j] - mx); sum += l[j]; }
    red[threadIdx.x] = sum; __syncthreads();
    for (int st = 128; st > 0; st >>= 1) {
        if (threadIdx.x < st) red[threadIdx.x] += red[threadIdx.x + st];
        __syncthreads();
    }
    const float inv = 1.0f / red[0];
    #pragma unroll
    for (int j = 0; j < 8; ++j) pr[threadIdx.x + j * 256] = f2bf(l[j] * inv);
}

// ---------------------------------------------------------------------------
// Head merge: o fp32 [B,H,T,Dh] -> bf16 [B,T,C]
// ---------------------------------------------------------------------------
__global__ __launch_bounds__(256) void transpose_o(const float* __restrict__ o,
                                                   unsigned short* __restrict__ obf) {
    const int idx = blockIdx.x * 256 + threadIdx.x;   // B*H*T*Dh
    const int d = idx & 127;
    const int t = (idx >> 7) & 255;
    const int h = (idx >> 15) & 15;
    const int b = idx >> 19;
    obf[((size_t)(b * NT + t) * NC) + h * NDH + d] = f2bf(o[idx]);
}

// ---------------------------------------------------------------------------
// Launch
// ---------------------------------------------------------------------------
extern "C" void kernel_launch(void* const* d_in, const int* in_sizes, int n_in,
                              void* d_out, int out_size, void* d_ws, size_t ws_size,
                              hipStream_t stream) {
    const float* x        = (const float*)d_in[0];
    const float* kcache   = (const float*)d_in[1];
    const float* vcache   = (const float*)d_in[2];
    const float* in_proj  = (const float*)d_in[3];
    const float* out_proj = (const float*)d_in[4];
    const float* norm1_a  = (const float*)d_in[5];
    const float* norm2_a  = (const float*)d_in[6];
    const float* w1       = (const float*)d_in[7];
    const float* w2       = (const float*)d_in[8];
    const int*   offp     = (const int*)d_in[9];
    float* out            = (float*)d_out;

    // Workspace bump allocator
    char* wp = (char*)d_ws;
    auto alloc = [&](size_t bytes) -> void* {
        void* r = (void*)wp;
        wp += (bytes + 255) & ~(size_t)255;
        return r;
    };
    unsigned short* wqkv_bf = (unsigned short*)alloc((size_t)3 * NC * NC * 2);
    unsigned short* wout_bf = (unsigned short*)alloc((size_t)NC * NC * 2);
    unsigned short* w1_bf   = (unsigned short*)alloc((size_t)NFF * NC * 2);
    unsigned short* w2_bf   = (unsigned short*)alloc((size_t)NC * NFF * 2);
    unsigned short* xn_bf   = (unsigned short*)alloc((size_t)NB * NT * NC * 2);
    float*          qkv     = (float*)         alloc((size_t)NB * NT * 3 * NC * 4);
    unsigned short* q_bf    = (unsigned short*)alloc((size_t)NB * NH * NT * NDH * 2);
    unsigned short* knew_bf = (unsigned short*)alloc((size_t)NB * NH * NT * NDH * 2);
    unsigned short* vnew_bf = (unsigned short*)alloc((size_t)NB * NH * NT * NDH * 2);
    unsigned short* kc_bf   = (unsigned short*)alloc((size_t)NB * NH * NCAP * NDH * 2);
    unsigned short* vT_bf   = (unsigned short*)alloc((size_t)NB * NH * NDH * NCAP * 2);
    unsigned short* attn_bf = (unsigned short*)alloc((size_t)NB * NH * NT * NCAP * 2);
    float*          o_f32   = (float*)         alloc((size_t)NB * NH * NT * NDH * 4);
    unsigned short* o_bf    = (unsigned short*)alloc((size_t)NB * NT * NC * 2);
    float*          x2      = (float*)         alloc((size_t)NB * NT * NC * 4);
    unsigned short* h_bf    = (unsigned short*)alloc((size_t)NB * NT * NC * 2);
    unsigned short* hm_bf   = (unsigned short*)alloc((size_t)NB * NT * NFF * 2);

    const int rows = NB * NT;            // 2048

    // 1) weights -> bf16
    cast_f32_bf16<<<4096, 256, 0, stream>>>(in_proj,  wqkv_bf, (long long)3 * NC * NC);
    cast_f32_bf16<<<4096, 256, 0, stream>>>(out_proj, wout_bf, (long long)NC * NC);
    cast_f32_bf16<<<4096, 256, 0, stream>>>(w1,       w1_bf,   (long long)NFF * NC);
    cast_f32_bf16<<<4096, 256, 0, stream>>>(w2,       w2_bf,   (long long)NC * NFF);

    // 2) RMSNorm1 -> bf16
    rmsnorm_bf16<<<rows, 256, 0, stream>>>(x, norm1_a, xn_bf);

    // 3) QKV projection: [2048,2048] x [6144,2048]^T -> fp32
    gemm_bf16<EPI_F32><<<dim3(3 * NC / 128, rows / 128, 1), 128, 0, stream>>>(
        xn_bf, wqkv_bf, qkv, nullptr, rows, 3 * NC, NC, 0, 0, 0);

    // 4) RoPE + head split
    rope_split<<<(NB * NH * NT * 64) / 256, 256, 0, stream>>>(qkv, offp, q_bf, knew_bf, vnew_bf);

    // 5) merge ring-buffer caches into bf16 K-cache and transposed V-cache
    merge_cache<<<((long long)NB * NH * NCAP * NDH) / 256, 256, 0, stream>>>(
        kcache, vcache, knew_bf, vnew_bf, offp, kc_bf, vT_bf);

    // 6) scores = q . K^T   (128 batches of 256x2048x128) -> bf16
    gemm_bf16<EPI_BF16><<<dim3(NCAP / 128, NT / 128, NB * NH), 128, 0, stream>>>(
        q_bf, kc_bf, attn_bf, nullptr, NT, NCAP, NDH,
        (long long)NT * NDH, (long long)NCAP * NDH, (long long)NT * NCAP);

    // 7) masked softmax (in place, bf16)
    softmax_mask<<<NB * NH * NT, 256, 0, stream>>>(attn_bf, offp);

    // 8) o = attn . V   (128 batches of 256x128x2048)
    gemm_bf16<EPI_F32><<<dim3(NDH / 128, NT / 128, NB * NH), 128, 0, stream>>>(
        attn_bf, vT_bf, o_f32, nullptr, NT, NDH, NCAP,
        (long long)NT * NCAP, (long long)NDH * NCAP, (long long)NT * NDH);

    // 9) head merge -> bf16 [B,T,C]
    transpose_o<<<(NB * NH * NT * NDH) / 256, 256, 0, stream>>>(o_f32, o_bf);

    // 10) out projection + residual: x2 = x + o @ Wout^T
    gemm_bf16<EPI_F32_RES><<<dim3(NC / 128, rows / 128, 1), 128, 0, stream>>>(
        o_bf, wout_bf, x2, x, rows, NC, NC, 0, 0, 0);

    // 11) RMSNorm2 -> bf16
    rmsnorm_bf16<<<rows, 256, 0, stream>>>(x2, norm2_a, h_bf);

    // 12) FFN up + exact GELU -> bf16
    gemm_bf16<EPI_GELU_BF16><<<dim3(NFF / 128, rows / 128, 1), 128, 0, stream>>>(
        h_bf, w1_bf, hm_bf, nullptr, rows, NFF, NC, 0, 0, 0);

    // 13) FFN down + residual -> d_out (fp32)
    gemm_bf16<EPI_F32_RES><<<dim3(NC / 128, rows / 128, 1), 128, 0, stream>>>(
        hm_bf, w2_bf, out, x2, rows, NC, NFF, 0, 0, 0);
}
